// TransducerJoiner_6133213299541
// MI455X (gfx1250) — compile-verified, matrix operand order checked by reading, not run on hardware
//
#include <hip/hip_runtime.h>
#include <hip/hip_bf16.h>

// Transducer joiner for MI455X (gfx1250).
// Roofline: 137.4 GFLOP vs 537 MB HBM output (~23us floor). bf16 WMMA (~5 PFLOPS
// dense) puts compute ~27us. The remaining hazard is L2 traffic (W + pred re-reads),
// so tile M=128 (2 t-values sharing the 64 pred rows) x V=128: B fragments feed two
// WMMAs each, halving both W and pred L2 streams vs a 64-row tile.

typedef __bf16 bf16;
typedef __attribute__((ext_vector_type(16))) bf16 v16bf;
typedef __attribute__((ext_vector_type(8)))  bf16 v8bf;
typedef __attribute__((ext_vector_type(4)))  bf16 v4bf;
typedef __attribute__((ext_vector_type(8)))  float v8f;

#define NB 8
#define TT 256
#define UU 64
#define CC 512
#define VV 1024
#define KCHUNK 256   // K staged per LDS pass (2 passes); 128 rows x 256 K bf16 = 64 KB

__device__ __forceinline__ float fast_tanh(float x) {
#if __has_builtin(__builtin_amdgcn_tanhf)
  return __builtin_amdgcn_tanhf(x);          // gfx1250 v_tanh_f32
#else
  float e = __expf(2.0f * x);
  return 1.0f - 2.0f / (e + 1.0f);
#endif
}

// Prologue: W (V x C f32) -> bf16 workspace (1 MB; L2-resident afterwards).
__global__ __launch_bounds__(256) void w_to_bf16(const float* __restrict__ W,
                                                 bf16* __restrict__ Wb) {
  int i = (blockIdx.x * 256 + threadIdx.x) * 4;
  float4 w = *reinterpret_cast<const float4*>(W + i);
  v4bf o = { (bf16)w.x, (bf16)w.y, (bf16)w.z, (bf16)w.w };
  *reinterpret_cast<v4bf*>(Wb + i) = o;
}

// Block: 2 t-values x 64 u (128 M-rows) x 128 vocab cols. 8 waves = 4(M-pairs) x 2(N).
// Wave: 2 M-frags x 4 N-frags = 8 accumulators; each B frag reused by 2 WMMAs.
__global__ __launch_bounds__(256) void joiner_wmma(
    const float* __restrict__ enc,   // (N,T,C)
    const float* __restrict__ pred,  // (N,U,C)
    const bf16*  __restrict__ Wb,    // (V,C) bf16
    const float* __restrict__ bias,  // (V)
    float* __restrict__ out) {       // (N,T,U,V)
  // 128 rows x 256 K bf16, XOR-swizzled at 16B chunks: chunk' = chunk ^ (row&15).
  // Row pitch 512B would alias all 16 fragment rows to one bank otherwise.
  __shared__ bf16 Alds[128 * KCHUNK];  // 64 KB

  const int tid = threadIdx.x;
  const int vb  = blockIdx.x;          // vocab tile -> cols vb*128 ..
  const int t0  = blockIdx.y * 2;      // two consecutive t values
  const int n   = blockIdx.z;

  const float* encR0 = enc  + (size_t)(n * TT + t0) * CC;   // t0 row (t0+1 = +CC)
  const float* predB = pred + (size_t)n * UU * CC;

  const int lane = tid & 31;
  const int w    = tid >> 5;
  const int wmp  = w & 3;              // M pair: rows 32*wmp .. 32*wmp+31
  const int wn   = w >> 2;             // N half: cols 64*wn ..
  const int l16  = lane & 15;
  const int hi   = lane >> 4;          // K-half select (16-bit A/B layout)
  const int vBase = vb * 128 + wn * 64;

  // Swizzle decomposition: chunk = 4j + hi + 2e, chunk^l16 = ((j^(l16>>2))<<2) | ((hi+2e)^(l16&3))
  const int mh  = l16 >> 2;
  const int lo0 = ((hi      ^ (l16 & 3)) << 3);   // halves offset, e=0 (K+0)
  const int lo1 = (((2 | hi) ^ (l16 & 3)) << 3);  // halves offset, e=1 (K+16)
  const bf16* aRow0 = &Alds[(32 * wmp + l16) * KCHUNK];
  const bf16* aRow1 = aRow0 + 16 * KCHUNK;

  union V16 { v16bf v; v8bf h[2]; };
  v8f acc[8] = {};

  // B column f == contiguous row of W (row-major): plain b128 loads.
  const bf16* wRow[4];
#pragma unroll
  for (int f = 0; f < 4; ++f)
    wRow[f] = Wb + (size_t)(vBase + f * 16 + l16) * CC + (hi << 3);

  for (int s = 0; s < 2; ++s) {
    const int kc = s * KCHUNK;
    if (s) __syncthreads();            // all waves done reading previous chunk

    // ---- Stage: A[m, kc..kc+255] = bf16(tanh(enc + pred)) ----
    for (int j = 0; j < 32; ++j) {
      int idx = j * 256 + tid;         // float4 index over 128 x 64 float4
      int m  = idx >> 6;               // 64 float4 per row
      int cl = (idx & 63) << 2;        // local column
      int u  = m & 63;
      const float* er = encR0 + (size_t)(m >> 6) * CC;   // t0 or t0+1
      float4 e = *reinterpret_cast<const float4*>(er + kc + cl);
      float4 p = *reinterpret_cast<const float4*>(predB + u * CC + kc + cl);
      v4bf a = { (bf16)fast_tanh(e.x + p.x), (bf16)fast_tanh(e.y + p.y),
                 (bf16)fast_tanh(e.z + p.z), (bf16)fast_tanh(e.w + p.w) };
      int sw = ((((cl >> 3) ^ (m & 15)) << 3) | (cl & 7));
      *reinterpret_cast<v4bf*>(&Alds[m * KCHUNK + sw]) = a;
    }
    __syncthreads();

    // ---- GEMM over this K chunk: 8 steps x 8 WMMA ----
    for (int j = 0; j < KCHUNK / 32; ++j) {
      const int kg = kc + j * 32;
      if (kg + 128 < CC)               // pull next W chunk into WGP caches
        __builtin_prefetch(wRow[0] + kg + 128, 0, 3);

      const int jx = ((j ^ mh) << 5);  // swizzled high chunk bits (halves)
      V16 a0, a1;
      a0.h[0] = *reinterpret_cast<const v8bf*>(aRow0 + jx + lo0);
      a0.h[1] = *reinterpret_cast<const v8bf*>(aRow0 + jx + lo1);
      a1.h[0] = *reinterpret_cast<const v8bf*>(aRow1 + jx + lo0);
      a1.h[1] = *reinterpret_cast<const v8bf*>(aRow1 + jx + lo1);

#pragma unroll
      for (int f = 0; f < 4; ++f) {
        V16 b;
        b.h[0] = *reinterpret_cast<const v8bf*>(wRow[f] + kg);
        b.h[1] = *reinterpret_cast<const v8bf*>(wRow[f] + kg + 16);
        acc[f]     = __builtin_amdgcn_wmma_f32_16x16x32_bf16(
            false, a0.v, false, b.v, (short)0, acc[f],     false, false);
        acc[4 + f] = __builtin_amdgcn_wmma_f32_16x16x32_bf16(
            false, a1.v, false, b.v, (short)0, acc[4 + f], false, false);
      }
    }
  }

  // ---- Epilogue: bias + f32 stores ----
  // C/D layout: VGPR r -> M = r (lanes 0-15) / r+8 (lanes 16-31); N = lane&15.
  const size_t outBase = (size_t)(n * TT + t0) * UU * VV;
  const int rofs = hi << 3;
#pragma unroll
  for (int f = 0; f < 4; ++f) {
    const int v = vBase + f * 16 + l16;
    const float bv = bias[v];
#pragma unroll
    for (int mf = 0; mf < 2; ++mf) {
#pragma unroll
      for (int r = 0; r < 8; ++r) {
        int m = 32 * wmp + 16 * mf + rofs + r;        // global tile row
        size_t o = outBase + (size_t)(m >> 6) * UU * VV   // t0 / t0+1
                 + (size_t)(m & 63) * VV + v;             // u, v
        out[o] = acc[4 * mf + f][r] + bv;
      }
    }
  }
}

extern "C" void kernel_launch(void* const* d_in, const int* in_sizes, int n_in,
                              void* d_out, int out_size, void* d_ws, size_t ws_size,
                              hipStream_t stream) {
  const float* enc  = (const float*)d_in[0];   // (N,T,C)
  const float* pred = (const float*)d_in[1];   // (N,U,C)
  const float* W    = (const float*)d_in[2];   // (V,C)
  const float* b    = (const float*)d_in[3];   // (V)
  float* out = (float*)d_out;                  // (N,T,U,V)

  bf16* Wb = (bf16*)d_ws;                      // needs V*C*2 = 1 MB scratch

  w_to_bf16<<<(VV * CC) / 1024, 256, 0, stream>>>(W, Wb);

  dim3 grid(VV / 128, TT / 2, NB);             // 8 x 128 x 8 = 8192 blocks
  joiner_wmma<<<grid, 256, 0, stream>>>(enc, pred, Wb, b, out);
}
